// LSTMModel_53412213293703
// MI455X (gfx1250) — compile-verified, matrix operand order checked by reading, not run on hardware
//
#include <hip/hip_runtime.h>
#include <hip/hip_bf16.h>

#define B_ 256
#define S_ 4096
#define D_ 64
#define H_ 128
#define G_ 512  // 4*H

typedef __attribute__((ext_vector_type(16))) __bf16 v16bf;
typedef __attribute__((ext_vector_type(8)))  float  v8f;

union BF16Frag {
  v16bf  v;
  __bf16 h[16];
  uint4  q[2];
};

__device__ __forceinline__ float fsigmoid(float x) {
  return 1.0f / (1.0f + __expf(-x));
}
__device__ __forceinline__ float ftanh_(float x) {
  // 2*sigmoid(2x)-1 : saturates correctly to +/-1 without NaN at large |x|
  return __builtin_fmaf(2.0f, fsigmoid(2.0f * x), -1.0f);
}

// ---------------------------------------------------------------------------
// Prologue: convert Wi [64x512] and Wh [128x512] from f32 to bf16 into d_ws.
// ws layout: [0 .. 32767] = Wi_bf16, [32768 .. 98303] = Wh_bf16
// ---------------------------------------------------------------------------
__global__ void cvt_weights_kernel(const float* __restrict__ Wi,
                                   const float* __restrict__ Wh,
                                   __bf16* __restrict__ wsb) {
  int i = blockIdx.x * blockDim.x + threadIdx.x;
  if (i < D_ * G_) wsb[i] = (__bf16)Wi[i];
  if (i < H_ * G_) wsb[D_ * G_ + i] = (__bf16)Wh[i];
}

// ---------------------------------------------------------------------------
// Fused LSTM scan. One block = 16 batch rows, 512 threads = 16 wave32.
// Each wave owns 2 N-tiles (32 gate columns) of the 512-wide gate dim.
// Weights live in registers as WMMA B fragments for the whole scan.
// ---------------------------------------------------------------------------
__global__ __launch_bounds__(512, 1)
void lstm_scan_kernel(const float* __restrict__ x,
                      const __bf16* __restrict__ Wib,
                      const __bf16* __restrict__ Whb,
                      const float* __restrict__ bh,
                      const float* __restrict__ Wo,
                      const float* __restrict__ bov,
                      float* __restrict__ out) {
  __shared__ float  lds_z[16 * G_];  // 32KB: gate pre-activations z[16][512]
  __shared__ __bf16 lds_h[16 * H_];  //  4KB: hidden state (bf16 A operand)
  __shared__ __bf16 lds_x[16 * D_];  //  2KB: staged x_t tile (bf16 A operand)

  const int tid   = threadIdx.x;
  const int lane  = tid & 31;
  const int wave  = tid >> 5;             // 0..15
  const int b0    = blockIdx.x * 16;      // batch tile base

  // --- WMMA fragment geometry (wave32, 16x16x32 bf16) ---
  const int bn    = lane & 15;            // N within tile (B/C/D lanes)
  const int am    = lane & 15;            // M within tile (A lanes)
  const int khalf = (lane >> 4) << 3;     // lanes 16..31 hold K+8 block
  const int nbase0 = (wave * 2 + 0) * 16; // this wave's gate columns
  const int nbase1 = (wave * 2 + 1) * 16;

  // --- Preload B fragments (weights) into registers: 12 frags/lane = 96 VGPR
  BF16Frag bWh[4][2];  // K = 0..127 in 4 steps of 32
  BF16Frag bWi[2][2];  // K = 0..63  in 2 steps of 32
#pragma unroll
  for (int k = 0; k < 4; ++k) {
#pragma unroll
    for (int nt = 0; nt < 2; ++nt) {
      const int n0 = (wave * 2 + nt) * 16;
      const int kb = k * 32 + khalf;
#pragma unroll
      for (int i = 0; i < 8; ++i) {
        bWh[k][nt].h[i]     = Whb[(kb + i)      * G_ + n0 + bn];
        bWh[k][nt].h[i + 8] = Whb[(kb + 16 + i) * G_ + n0 + bn];
      }
    }
  }
#pragma unroll
  for (int k = 0; k < 2; ++k) {
#pragma unroll
    for (int nt = 0; nt < 2; ++nt) {
      const int n0 = (wave * 2 + nt) * 16;
      const int kb = k * 32 + khalf;
#pragma unroll
      for (int i = 0; i < 8; ++i) {
        bWi[k][nt].h[i]     = Wib[(kb + i)      * G_ + n0 + bn];
        bWi[k][nt].h[i + 8] = Wib[(kb + 16 + i) * G_ + n0 + bn];
      }
    }
  }

  // --- Gate-phase mapping: thread -> column gj, rows gm0..gm0+3 ---
  const int gj  = tid & 127;
  const int gm0 = (tid >> 7) * 4;
  const float bi_ = bh[gj];
  const float bf_ = bh[gj + 128];
  const float bg_ = bh[gj + 256];
  const float bo_ = bh[gj + 384];
  float creg[4] = {0.f, 0.f, 0.f, 0.f};  // cell state, persistent in VGPRs
  float hreg[4] = {0.f, 0.f, 0.f, 0.f};

  // --- x staging mapping: thread -> (row xm, 2 elements at xd) ---
  const int xm = tid >> 5;
  const int xd = (tid & 31) * 2;
  const float* xrow = x + (size_t)(b0 + xm) * S_ * D_ + xd;

  // zero initial hidden state
#pragma unroll
  for (int r = 0; r < 4; ++r)
    lds_h[(gm0 + r) * H_ + gj] = (__bf16)0.0f;

  for (int t = 0; t < S_; ++t) {
    // ---- phase A: stage x_t tile [16 x 64] f32 -> bf16 in LDS ----
    const float* xp = xrow + (size_t)t * D_;
    float2 xv = *(const float2*)xp;
    lds_x[xm * D_ + xd]     = (__bf16)xv.x;
    lds_x[xm * D_ + xd + 1] = (__bf16)xv.y;
    if (t + 8 < S_) __builtin_prefetch((const void*)(xp + 8 * D_), 0, 0);
    __syncthreads();  // lds_x ready; prev step's lds_h visible

    // ---- phase B: z = x_t@Wi + h@Wh via bf16 WMMA, f32 accumulate ----
    v8f acc0 = {0.f, 0.f, 0.f, 0.f, 0.f, 0.f, 0.f, 0.f};
    v8f acc1 = {0.f, 0.f, 0.f, 0.f, 0.f, 0.f, 0.f, 0.f};
#pragma unroll
    for (int k = 0; k < 2; ++k) {  // input path, K = 0..63
      BF16Frag a;
      const __bf16* ap = &lds_x[am * D_ + k * 32 + khalf];
      a.q[0] = *(const uint4*)ap;
      a.q[1] = *(const uint4*)(ap + 16);
      acc0 = __builtin_amdgcn_wmma_f32_16x16x32_bf16(
          false, a.v, false, bWi[k][0].v, (short)0, acc0, false, false);
      acc1 = __builtin_amdgcn_wmma_f32_16x16x32_bf16(
          false, a.v, false, bWi[k][1].v, (short)0, acc1, false, false);
    }
#pragma unroll
    for (int k = 0; k < 4; ++k) {  // recurrent path, K = 0..127
      BF16Frag a;
      const __bf16* ap = &lds_h[am * H_ + k * 32 + khalf];
      a.q[0] = *(const uint4*)ap;
      a.q[1] = *(const uint4*)(ap + 16);
      acc0 = __builtin_amdgcn_wmma_f32_16x16x32_bf16(
          false, a.v, false, bWh[k][0].v, (short)0, acc0, false, false);
      acc1 = __builtin_amdgcn_wmma_f32_16x16x32_bf16(
          false, a.v, false, bWh[k][1].v, (short)0, acc1, false, false);
    }
    // spill z tiles to LDS (C/D layout: VGPR r -> row r (+8 for hi lanes))
    const int zr = (lane >> 4) << 3;
#pragma unroll
    for (int r = 0; r < 8; ++r) {
      lds_z[(zr + r) * G_ + nbase0 + bn] = acc0[r];
      lds_z[(zr + r) * G_ + nbase1 + bn] = acc1[r];
    }
    __syncthreads();  // z complete

    // ---- phase C: gates + state update (f32), h -> bf16 for next step ----
#pragma unroll
    for (int r = 0; r < 4; ++r) {
      const int m = gm0 + r;
      const float zi = lds_z[m * G_ + gj]       + bi_;
      const float zf = lds_z[m * G_ + 128 + gj] + bf_;
      const float zg = lds_z[m * G_ + 256 + gj] + bg_;
      const float zo = lds_z[m * G_ + 384 + gj] + bo_;
      const float ig = fsigmoid(zi);
      const float fg = fsigmoid(zf);
      const float gg = ftanh_(zg);
      const float og = fsigmoid(zo);
      const float cn = __builtin_fmaf(fg, creg[r], ig * gg);
      creg[r] = cn;
      const float hn = og * ftanh_(cn);
      hreg[r] = hn;
      lds_h[m * H_ + gj] = (__bf16)hn;
    }
  }

  // ---- epilogue: out = h_last @ Wo + bo (full f32 path) ----
  __syncthreads();                // everyone done reading lds_z
  float* hF = lds_z;              // reuse z buffer as f32 h_last [16][128]
#pragma unroll
  for (int r = 0; r < 4; ++r) hF[(gm0 + r) * H_ + gj] = hreg[r];
  __syncthreads();
  if (tid < 16) {
    float acc = bov[0];
    for (int j = 0; j < H_; ++j) acc += hF[tid * H_ + j] * Wo[j];
    out[b0 + tid] = acc;
  }
}

// ---------------------------------------------------------------------------
extern "C" void kernel_launch(void* const* d_in, const int* in_sizes, int n_in,
                              void* d_out, int out_size, void* d_ws, size_t ws_size,
                              hipStream_t stream) {
  const float* x  = (const float*)d_in[0];
  const float* Wi = (const float*)d_in[1];
  const float* Wh = (const float*)d_in[2];
  const float* bh = (const float*)d_in[3];
  const float* Wo = (const float*)d_in[4];
  const float* bo = (const float*)d_in[5];
  float* out = (float*)d_out;
  __bf16* wsb = (__bf16*)d_ws;  // needs (64+128)*512*2 = 192KB of scratch

  cvt_weights_kernel<<<(H_ * G_ + 255) / 256, 256, 0, stream>>>(Wi, Wh, wsb);
  lstm_scan_kernel<<<B_ / 16, 512, 0, stream>>>(
      x, /*Wi_bf16=*/wsb, /*Wh_bf16=*/wsb + D_ * G_, bh, Wo, bo, out);
}
